// Decoder_single_LSTM_78718160601583
// MI455X (gfx1250) — compile-verified, matrix operand order checked by reading, not run on hardware
//
#include <hip/hip_runtime.h>
#include <hip/hip_bf16.h>

// ---------------------------------------------------------------------------
// Decoder (PreNet -> LSTM(768) -> proj) for MI455X / gfx1250, wave32 + WMMA.
// All GEMMs use v_wmma_f32_16x16x32_bf16 (bf16 in, f32 accumulate).
//   A-fragments  : 2x contiguous 16B chunks per lane (global/ds _b128)
//   B-fragments  : weights pre-packed to fragment-native layout, 32B/lane
//   xgemm        : 4 N-tiles per wave -> A-fragment register reuse x4
// Sequential LSTM: persistent 96-WG kernel, grid barrier per timestep,
// w_hh slice fragment-packed in LDS, next-step xg prefetch.
// ---------------------------------------------------------------------------

typedef __bf16 bf16;
typedef __attribute__((ext_vector_type(16))) __bf16 v16bf;
typedef __attribute__((ext_vector_type(8)))  __bf16 v8bf;
typedef __attribute__((ext_vector_type(2)))  __bf16 v2bf;
typedef __attribute__((ext_vector_type(8)))  float  v8f;

#define B_SZ   32
#define T_SZ   2048
#define ROWS   (B_SZ * T_SZ)      // 65536
#define H_SZ   768
#define G_SZ   (4 * H_SZ)         // 3072
#define LSTM_WGS 96               // 768 hidden cols / 8 per WG

// --- WMMA fragment index map (CDNA5 ISA 7.12.2, 16-bit A 16x32) -------------
// lane L (half = L>>4), element e  ->  k = (e<8 ? e : 8+e) + 8*half
static __device__ __forceinline__ int frag_k(int e, int half) {
  return (e < 8 ? e : 8 + e) + (half << 3);
}

#define CAT16 0,1,2,3,4,5,6,7,8,9,10,11,12,13,14,15

// A fragment: 16x32 tile of row-major bf16 matrix -> two 16B vector loads.
static __device__ __forceinline__ v16bf load_a_bf16(const bf16* A, int lda, int m0, int k0) {
  const int lane = threadIdx.x & 31;
  const bf16* p = A + (size_t)(m0 + (lane & 15)) * lda + k0 + ((lane >> 4) << 3);
  const v8bf lo = *(const v8bf*)p;
  const v8bf hi = *(const v8bf*)(p + 16);
  return __builtin_shufflevector(lo, hi, CAT16);
}

// B fragment from fragment-packed weights: [kt][nt][lane][16] -> 32B/lane.
static __device__ __forceinline__ v16bf load_b_packed(const bf16* P, int NT, int kt, int nt) {
  const int lane = threadIdx.x & 31;
  const bf16* p = P + ((((size_t)kt * NT + nt) << 5) + lane) * 16;
  const v8bf lo = *(const v8bf*)p;
  const v8bf hi = *(const v8bf*)(p + 8);
  return __builtin_shufflevector(lo, hi, CAT16);
}

static __device__ __forceinline__ v8f wmma_bf16(v16bf a, v16bf b, v8f c) {
  return __builtin_amdgcn_wmma_f32_16x16x32_bf16(false, a, false, b, (short)0, c, false, false);
}

static __device__ __forceinline__ float sigm(float x) {
  return 1.0f / (1.0f + __expf(-x));
}

// ---------------------------------------------------------------------------
// Elementwise helper kernels
// ---------------------------------------------------------------------------
__global__ void cvt_bf16_kernel(const float* __restrict__ s, bf16* __restrict__ d, int n) {
  for (int i = blockIdx.x * blockDim.x + threadIdx.x; i < n; i += gridDim.x * blockDim.x)
    d[i] = (bf16)s[i];
}

// pack row-major f32 weight [K][N] into WMMA-B fragment layout bf16:
// out[(((kt*NT + nt)*32 + lane)*16 + e] = w[(kt*32 + frag_k(e,half))*N + nt*16 + lane%16]
__global__ void pack_w_kernel(const float* __restrict__ w, bf16* __restrict__ out, int K, int N) {
  const int NT = N >> 4;
  const int total = K * N;
  for (int i = blockIdx.x * blockDim.x + threadIdx.x; i < total; i += gridDim.x * blockDim.x) {
    const int e    = i & 15;
    const int lane = (i >> 4) & 31;
    const int r    = i >> 9;
    const int nt   = r % NT;
    const int kt   = r / NT;
    const int half = lane >> 4;
    const int col  = (nt << 4) + (lane & 15);
    const int k    = (kt << 5) + frag_k(e, half);
    out[i] = (bf16)w[(size_t)k * N + col];
  }
}

__global__ void bias_add_kernel(const float* __restrict__ a, const float* __restrict__ b,
                                float* __restrict__ o, int n) {
  for (int i = blockIdx.x * blockDim.x + threadIdx.x; i < n; i += gridDim.x * blockDim.x)
    o[i] = a[i] + b[i];
}

__global__ void zero_u32_kernel(unsigned* __restrict__ p, int n) {
  for (int i = blockIdx.x * blockDim.x + threadIdx.x; i < n; i += gridDim.x * blockDim.x)
    p[i] = 0u;
}

// x [ROWS,512] f32  -> inp[:, 0:512] bf16 (concat buffer, ld = 768)
__global__ void pack_x_kernel(const float* __restrict__ x, bf16* __restrict__ inp, int n) {
  for (int i = blockIdx.x * blockDim.x + threadIdx.x; i < n; i += gridDim.x * blockDim.x) {
    const int r = i >> 9, c = i & 511;
    inp[(size_t)r * 768 + c] = (bf16)x[i];
  }
}

// ---------------------------------------------------------------------------
// PreNet: relu(relu(mels@w1+b1)@w2+b2) -> inp[:, 512:768]   (16 rows / WG)
// ---------------------------------------------------------------------------
__global__ void __launch_bounds__(256) prenet_kernel(
    const bf16* __restrict__ melsb,   // [ROWS][128] bf16
    const bf16* __restrict__ w1p, const float* __restrict__ b1,
    const bf16* __restrict__ w2p, const float* __restrict__ b2,
    bf16* __restrict__ inp) {
  __shared__ bf16 m1[16 * 256];
  const int wave = threadIdx.x >> 5, lane = threadIdx.x & 31;
  const int m0 = blockIdx.x * 16;
  const int col = lane & 15;
  const int rbase = (lane >> 4) << 3;

  // layer 1: [16,128] x [128,256]
  for (int nt = wave; nt < 16; nt += 8) {
    v8f acc;
    const float bv = b1[(nt << 4) + col];
#pragma unroll
    for (int v = 0; v < 8; ++v) acc[v] = bv;
#pragma unroll
    for (int ks = 0; ks < 4; ++ks) {
      v16bf a = load_a_bf16(melsb, 128, m0, ks * 32);
      v16bf b = load_b_packed(w1p, 16, ks, nt);
      acc = wmma_bf16(a, b, acc);
    }
#pragma unroll
    for (int v = 0; v < 8; ++v) {
      const int row = v + rbase;
      const float r = acc[v] > 0.f ? acc[v] : 0.f;
      m1[row * 256 + (nt << 4) + col] = (bf16)r;
    }
  }
  __syncthreads();

  // layer 2: [16,256] x [256,256]
  for (int nt = wave; nt < 16; nt += 8) {
    v8f acc;
    const float bv = b2[(nt << 4) + col];
#pragma unroll
    for (int v = 0; v < 8; ++v) acc[v] = bv;
#pragma unroll
    for (int ks = 0; ks < 8; ++ks) {
      v16bf a = load_a_bf16(m1, 256, 0, ks * 32);
      v16bf b = load_b_packed(w2p, 16, ks, nt);
      acc = wmma_bf16(a, b, acc);
    }
#pragma unroll
    for (int v = 0; v < 8; ++v) {
      const int row = m0 + v + rbase;
      const float r = acc[v] > 0.f ? acc[v] : 0.f;
      inp[(size_t)row * 768 + 512 + (nt << 4) + col] = (bf16)r;
    }
  }
}

// ---------------------------------------------------------------------------
// xg = inp @ w_ih + (b_ih+b_hh) : [65536,768] x [768,3072] -> f32
// 4 N-tiles per wave (A-fragment reused x4 from registers), 24 k-steps.
// ---------------------------------------------------------------------------
__global__ void __launch_bounds__(256) xgemm_kernel(
    const bf16* __restrict__ inp, const bf16* __restrict__ wihp,
    const float* __restrict__ bias, float* __restrict__ xg) {
  const int wave = threadIdx.x >> 5, lane = threadIdx.x & 31;
  const int gid = blockIdx.x * 8 + wave;    // 4096 mt x 48 nt-groups
  const int mt = gid / 48, ntg = gid % 48;
  const int m0 = mt * 16, nt0 = ntg * 4;
  const int col = lane & 15, rbase = (lane >> 4) << 3;
  v8f acc[4];
#pragma unroll
  for (int j = 0; j < 4; ++j) {
    const float bv = bias[((nt0 + j) << 4) + col];
#pragma unroll
    for (int v = 0; v < 8; ++v) acc[j][v] = bv;
  }
#pragma unroll 2
  for (int ks = 0; ks < 24; ++ks) {
    v16bf a = load_a_bf16(inp, 768, m0, ks * 32);
#pragma unroll
    for (int j = 0; j < 4; ++j) {
      v16bf b = load_b_packed(wihp, 192, ks, nt0 + j);
      acc[j] = wmma_bf16(a, b, acc[j]);
    }
  }
#pragma unroll
  for (int j = 0; j < 4; ++j)
#pragma unroll
    for (int v = 0; v < 8; ++v) {
      const int row = m0 + v + rbase;
      xg[(size_t)row * 3072 + ((nt0 + j) << 4) + col] = acc[j][v];
    }
}

// ---------------------------------------------------------------------------
// Persistent LSTM: 96 WGs x 128 threads. WG owns h-cols [8*wg, 8*wg+8);
// its 32 gate columns (i|f|g|o) form two 16-wide WMMA N-tiles.
// w_hh slice fragment-packed into LDS (48KB): [ks(24)][nt(2)][lane][16].
// Per step: [32,768] x [768,32] via 4 waves x 24 wmma (2 acc chains),
// gate math (paired-column b32 stores), grid barrier, next-step xg prefetch.
// ---------------------------------------------------------------------------
__global__ void __launch_bounds__(128) lstm_kernel(
    const float* __restrict__ xg,     // [B][T][3072]
    const float* __restrict__ w_hh,   // [768][3072] f32
    bf16* __restrict__ hbuf,          // [2][32][768] bf16 (double buffer)
    bf16* __restrict__ hs,            // [B][T][768] bf16
    unsigned* __restrict__ syncc) {
  __shared__ bf16 wp[24 * 2 * 32 * 16];  // fragment-packed w_hh slice, 48KB
  __shared__ float gatesS[32 * 32];      // [batch][local gate col]
  __shared__ float cS[32 * 8];           // [batch][local h col]

  const int tid = threadIdx.x, wave = tid >> 5, lane = tid & 31;
  const int colBase = blockIdx.x * 8;

  // pack this WG's w_hh slice into LDS fragment layout
  for (int i = tid; i < 24 * 2 * 32 * 16; i += 128) {
    const int e    = i & 15;
    const int ln   = (i >> 4) & 31;
    const int r    = i >> 9;
    const int nt   = r & 1;
    const int ks   = r >> 1;
    const int lc   = (nt << 4) + (ln & 15);      // local gate col 0..31
    const int gate = lc >> 3, c = lc & 7;
    const int k    = (ks << 5) + frag_k(e, ln >> 4);
    wp[i] = (bf16)w_hh[(size_t)k * 3072 + gate * 768 + colBase + c];
  }
  for (int i = tid; i < 32 * 8; i += 128) cS[i] = 0.f;
  __syncthreads();

  const int mt = wave >> 1, nt = wave & 1;
  const int m0 = mt * 16;
  const int col = lane & 15, rbase = (lane >> 4) << 3;
  const int lc = (nt << 4) + col;                        // local gate col
  const int gcol = (lc >> 3) * 768 + colBase + (lc & 7); // global gate col
  // state-update mapping: thread owns batch b, column pair cp..cp+1
  const int ub = tid >> 2, ucp = (tid & 3) << 1;

  for (int t = 0; t < T_SZ; ++t) {
    const bf16* hcur = hbuf + (size_t)(t & 1) * (32 * 768);
    bf16* hnxt = hbuf + (size_t)((t + 1) & 1) * (32 * 768);

    // two interleaved accumulator chains to halve dependent-WMMA depth
    v8f acc0, acc1;
#pragma unroll
    for (int v = 0; v < 8; ++v) {
      const int b = m0 + v + rbase;                      // batch row
      acc0[v] = xg[((size_t)b * T_SZ + t) * 3072 + gcol];
      acc1[v] = 0.f;
    }
    // prefetch next timestep's gate slice while this step computes
    if (t + 1 < T_SZ) {
#pragma unroll
      for (int v = 0; v < 8; ++v) {
        const int b = m0 + v + rbase;
        __builtin_prefetch(&xg[((size_t)b * T_SZ + (t + 1)) * 3072 + gcol], 0, 3);
      }
    }
#pragma unroll 4
    for (int ks = 0; ks < 24; ks += 2) {
      v16bf a0 = load_a_bf16(hcur, 768, m0, ks * 32);
      v16bf b0 = load_b_packed(wp, 2, ks, nt);
      v16bf a1 = load_a_bf16(hcur, 768, m0, (ks + 1) * 32);
      v16bf b1 = load_b_packed(wp, 2, ks + 1, nt);
      acc0 = wmma_bf16(a0, b0, acc0);
      acc1 = wmma_bf16(a1, b1, acc1);
    }
#pragma unroll
    for (int v = 0; v < 8; ++v) {
      const int b = m0 + v + rbase;
      gatesS[b * 32 + lc] = acc0[v] + acc1[v];
    }
    __syncthreads();

    // gate nonlinearity + state update: each thread owns (b, cp..cp+1)
    {
      v2bf hv;
#pragma unroll
      for (int j = 0; j < 2; ++j) {
        const int c = ucp + j;
        const float gi = sigm(gatesS[ub * 32 + 0  + c]);
        const float gf = sigm(gatesS[ub * 32 + 8  + c]);
        const float gg = tanhf(gatesS[ub * 32 + 16 + c]);
        const float go = sigm(gatesS[ub * 32 + 24 + c]);
        const float cn = gf * cS[ub * 8 + c] + gi * gg;
        cS[ub * 8 + c] = cn;
        hv[j] = (bf16)(go * tanhf(cn));
      }
      *(v2bf*)(hnxt + ub * 768 + colBase + ucp) = hv;
      *(v2bf*)(hs + ((size_t)ub * T_SZ + t) * 768 + colBase + ucp) = hv;
    }

    // grid-wide barrier (monotonic counter; h double-buffered)
    __threadfence();
    __syncthreads();
    if (tid == 0) {
      atomicAdd(syncc, 1u);
      const unsigned target = (unsigned)LSTM_WGS * (unsigned)(t + 1);
      volatile unsigned* vs = (volatile unsigned*)syncc;
      while (*vs < target) { __builtin_amdgcn_s_sleep(2); }
      __threadfence();
    }
    __syncthreads();
  }
}

// ---------------------------------------------------------------------------
// out = hs @ w_proj : [65536,768] x [768,128] -> f32 (w_proj fragment-packed)
// ---------------------------------------------------------------------------
__global__ void __launch_bounds__(256) proj_kernel(
    const bf16* __restrict__ hs, const bf16* __restrict__ wpp, float* __restrict__ out) {
  const int wave = threadIdx.x >> 5, lane = threadIdx.x & 31;
  const int gid = blockIdx.x * 8 + wave;    // 4096 mt x 2 nt-groups
  const int mt = gid >> 1, ntg = gid & 1;
  const int m0 = mt * 16, nt0 = ntg * 4;
  const int col = lane & 15, rbase = (lane >> 4) << 3;
  v8f acc[4];
#pragma unroll
  for (int j = 0; j < 4; ++j)
#pragma unroll
    for (int v = 0; v < 8; ++v) acc[j][v] = 0.f;
#pragma unroll 2
  for (int ks = 0; ks < 24; ++ks) {
    v16bf a = load_a_bf16(hs, 768, m0, ks * 32);
#pragma unroll
    for (int j = 0; j < 4; ++j) {
      v16bf b = load_b_packed(wpp, 8, ks, nt0 + j);
      acc[j] = wmma_bf16(a, b, acc[j]);
    }
  }
#pragma unroll
  for (int j = 0; j < 4; ++j)
#pragma unroll
    for (int v = 0; v < 8; ++v) {
      const int row = m0 + v + rbase;
      out[(size_t)row * 128 + ((nt0 + j) << 4) + col] = acc[j][v];
    }
}

// ---------------------------------------------------------------------------
extern "C" void kernel_launch(void* const* d_in, const int* in_sizes, int n_in,
                              void* d_out, int out_size, void* d_ws, size_t ws_size,
                              hipStream_t stream) {
  (void)in_sizes; (void)n_in; (void)out_size; (void)ws_size;
  const float* x      = (const float*)d_in[0];
  const float* mels   = (const float*)d_in[1];
  const float* w1     = (const float*)d_in[2];
  const float* b1     = (const float*)d_in[3];
  const float* w2     = (const float*)d_in[4];
  const float* b2     = (const float*)d_in[5];
  const float* w_ih   = (const float*)d_in[6];
  const float* w_hh   = (const float*)d_in[7];
  const float* b_ih   = (const float*)d_in[8];
  const float* b_hh   = (const float*)d_in[9];
  const float* w_proj = (const float*)d_in[10];
  float* out = (float*)d_out;

  // workspace carve-out (~1.04 GB)
  char* p = (char*)d_ws;
  auto carve = [&](size_t bytes) {
    char* r = p;
    p += (bytes + 255) & ~(size_t)255;
    return (void*)r;
  };
  bf16*     inp   = (bf16*)    carve((size_t)ROWS * 768 * sizeof(bf16));
  float*    xg    = (float*)   carve((size_t)ROWS * G_SZ * sizeof(float));
  bf16*     hs    = (bf16*)    carve((size_t)ROWS * 768 * sizeof(bf16));
  bf16*     melsb = (bf16*)    carve((size_t)ROWS * 128 * sizeof(bf16));
  bf16*     hbuf  = (bf16*)    carve((size_t)2 * 32 * 768 * sizeof(bf16));
  bf16*     w1p   = (bf16*)    carve((size_t)128 * 256 * sizeof(bf16));
  bf16*     w2p   = (bf16*)    carve((size_t)256 * 256 * sizeof(bf16));
  bf16*     wihp  = (bf16*)    carve((size_t)768 * G_SZ * sizeof(bf16));
  bf16*     wpp   = (bf16*)    carve((size_t)768 * 128 * sizeof(bf16));
  float*    bias  = (float*)   carve((size_t)G_SZ * sizeof(float));
  unsigned* syncc = (unsigned*)carve(256);

  // 1) weight packing (f32 -> bf16, fragment-native layout)
  pack_w_kernel<<<256, 256, 0, stream>>>(w1, w1p, 128, 256);
  pack_w_kernel<<<256, 256, 0, stream>>>(w2, w2p, 256, 256);
  pack_w_kernel<<<2048, 256, 0, stream>>>(w_ih, wihp, 768, G_SZ);
  pack_w_kernel<<<256, 256, 0, stream>>>(w_proj, wpp, 768, 128);
  // 2) mels f32 -> bf16, fused bias, zero-init (h buffer + barrier counter)
  cvt_bf16_kernel<<<4096, 256, 0, stream>>>(mels, melsb, ROWS * 128);
  bias_add_kernel<<<12, 256, 0, stream>>>(b_ih, b_hh, bias, G_SZ);
  zero_u32_kernel<<<96, 256, 0, stream>>>((unsigned*)hbuf, 2 * 32 * 768 / 2);
  zero_u32_kernel<<<1, 64, 0, stream>>>(syncc, 64);
  // 3) pack x into concat input (cols 0..511)
  pack_x_kernel<<<4096, 256, 0, stream>>>(x, inp, ROWS * 512);
  // 4) PreNet -> concat input (cols 512..767)
  prenet_kernel<<<ROWS / 16, 256, 0, stream>>>(melsb, w1p, b1, w2p, b2, inp);
  // 5) big input GEMM: xg = inp @ w_ih + bias
  xgemm_kernel<<<(ROWS / 16) * (G_SZ / 64) / 8, 256, 0, stream>>>(inp, wihp, bias, xg);
  // 6) persistent sequential LSTM
  lstm_kernel<<<LSTM_WGS, 128, 0, stream>>>(xg, w_hh, hbuf, hs, syncc);
  // 7) output projection
  proj_kernel<<<(ROWS / 16) * (128 / 64) / 8, 256, 0, stream>>>(hs, wpp, out);
}
